// VAE_84653805404711
// MI455X (gfx1250) — compile-verified, matrix operand order checked by reading, not run on hardware
//
#include <hip/hip_runtime.h>
#include <hip/hip_bf16.h>
#include <stddef.h>

// Problem dims
#define D_   256
#define H_   16
#define DZ_  64
#define L_   4
#define DFF_ 1024
#define B_   8
#define N_   64
#define BN_  (B_*N_)                 // 512 node rows
#define BNN_ (B_*N_*N_)              // 32768 edge rows
#define XE_  (BN_*D_)                // 131072
#define EE_  ((size_t)BNN_*D_)       // 8388608
#define ATTE_ (BNN_*H_)              // 524288

typedef __attribute__((ext_vector_type(16))) __bf16 v16bf;
typedef __attribute__((ext_vector_type(8)))  __bf16 v8bf;
typedef __attribute__((ext_vector_type(8)))  float  v8f;

// ---------- bf16 helpers ----------
__device__ __forceinline__ unsigned bfr(float f) {          // round-to-nearest-even, result in high 16
  unsigned u = __builtin_bit_cast(unsigned, f);
  return u + 0x7FFFu + ((u >> 16) & 1u);
}
// pack two floats -> two bf16 in one u32 (f0 low, f1 high) with a single v_perm_b32
__device__ __forceinline__ unsigned pack2bf(float f0, float f1) {
  return __builtin_amdgcn_perm(bfr(f1), bfr(f0), 0x07060302u);
}
__device__ __forceinline__ __bf16 f2bf(float f) {
  unsigned short s = (unsigned short)(bfr(f) >> 16);
  __bf16 r;
  __builtin_memcpy(&r, &s, sizeof(r));
  return r;
}

// Fragment (16 bf16 per lane): elems 0..7 at p[0..7], elems 8..15 at p[16..23]
// (A: lane&15 = row, lane>>4 = K-half; B from transposed weights: lane&15 = col)
__device__ __forceinline__ v16bf load_frag16(const __bf16* __restrict__ p, bool valid) {
  if (valid) {
    const v8bf* q = (const v8bf*)p;          // 16B aligned by construction
    v8bf lo = q[0];
    v8bf hi = q[2];                          // +16 elements
    return __builtin_shufflevector(lo, hi, 0,1,2,3,4,5,6,7,8,9,10,11,12,13,14,15);
  }
  v16bf a;
#pragma unroll
  for (int e = 0; e < 16; ++e) a[e] = (__bf16)0.0f;
  return a;
}

// ---------- workhorse bf16 WMMA GEMM: C = A(MxK) @ W(KxN) + bias [+ R] ----------
// A: bf16 row-major MxK.  Wt: bf16 TRANSPOSED (N rows x K cols).
// grid = (N/64, ceil(M/64)), block = 128 (4 waves); wave handles 16 rows x 64 cols.
template<bool RESID>
__global__ __launch_bounds__(128) void gemm_bf16t(const __bf16* __restrict__ Ab,
                                                  const __bf16* __restrict__ Wt,
                                                  const float* __restrict__ bias,
                                                  const float* __restrict__ R,
                                                  float* __restrict__ C,
                                                  int M, int K, int Nn) {
  const int lane = threadIdx.x & 31;
  const int wave = threadIdx.x >> 5;
  const int l15  = lane & 15;
  const int h    = lane >> 4;
  const int m0   = blockIdx.y * 64 + wave * 16;
  const int n0   = blockIdx.x * 64;
  const int m    = m0 + l15;
  const bool mvalid = (m < M);

  v8f acc[4];
#pragma unroll
  for (int t = 0; t < 4; ++t)
#pragma unroll
    for (int r = 0; r < 8; ++r) acc[t][r] = 0.f;

  for (int kb = 0; kb < K; kb += 32) {
    v16bf a = load_frag16(Ab + (size_t)m * K + kb + 8*h, mvalid);
#pragma unroll
    for (int t = 0; t < 4; ++t) {
      v16bf b = load_frag16(Wt + (size_t)(n0 + t*16 + l15) * K + kb + 8*h, true);
      acc[t] = __builtin_amdgcn_wmma_f32_16x16x32_bf16(false, a, false, b,
                                                       (short)0, acc[t], false, false);
    }
  }
#pragma unroll
  for (int t = 0; t < 4; ++t) {
    const int col = n0 + t*16 + l15;
#pragma unroll
    for (int r = 0; r < 8; ++r) {
      const int row = m0 + h*8 + r;
      if (row < M) {
        float v = acc[t][r] + bias[col];
        if constexpr (RESID) v += R[(size_t)row * Nn + col];
        C[(size_t)row * Nn + col] = v;
      }
    }
  }
}

// ---------- fused FFN: X += relu(Hb@W1 + b1) @ W2 + b2   (256 -> 1024 -> 256) ----------
// Hb: bf16 Mx256.  W1t: bf16 transposed 1024x256.  W2t: bf16 transposed 256x1024.
__global__ __launch_bounds__(256) void ffn_fused_bf(const __bf16* __restrict__ Hb,
                                                    const __bf16* __restrict__ W1t,
                                                    const float* __restrict__ b1,
                                                    const __bf16* __restrict__ W2t,
                                                    const float* __restrict__ b2,
                                                    float* __restrict__ X, int M) {
  __shared__ __attribute__((aligned(16))) __bf16 hid[16 * DFF_];   // 32 KB
  const int lane = threadIdx.x & 31;
  const int wave = threadIdx.x >> 5;
  const int l15  = lane & 15;
  const int h    = lane >> 4;
  const int m0   = blockIdx.x * 16;
  const int m    = m0 + l15;
  const bool mvalid = (m < M);

  // Phase 1: hidden cols [wave*128, wave*128+128)
  v8f acc[8];
#pragma unroll
  for (int t = 0; t < 8; ++t)
#pragma unroll
    for (int r = 0; r < 8; ++r) acc[t][r] = 0.f;

  for (int kb = 0; kb < D_; kb += 32) {
    v16bf a = load_frag16(Hb + (size_t)m * D_ + kb + 8*h, mvalid);
#pragma unroll
    for (int t = 0; t < 8; ++t) {
      v16bf b = load_frag16(W1t + (size_t)(wave*128 + t*16 + l15) * D_ + kb + 8*h, true);
      acc[t] = __builtin_amdgcn_wmma_f32_16x16x32_bf16(false, a, false, b,
                                                       (short)0, acc[t], false, false);
    }
  }
#pragma unroll
  for (int t = 0; t < 8; ++t) {
    const int col = wave*128 + t*16 + l15;
#pragma unroll
    for (int r = 0; r < 8; ++r) {
      const int rr = h*8 + r;
      hid[rr * DFF_ + col] = f2bf(fmaxf(acc[t][r] + b1[col], 0.f));
    }
  }
  __syncthreads();

  // Phase 2: output cols [wave*32, wave*32+32), K over 1024 from LDS
  v8f acc2[2];
#pragma unroll
  for (int t = 0; t < 2; ++t)
#pragma unroll
    for (int r = 0; r < 8; ++r) acc2[t][r] = 0.f;

  for (int kb = 0; kb < DFF_; kb += 32) {
    const v8bf* hp = (const v8bf*)(hid + l15 * DFF_ + kb + 8*h);
    v8bf lo = hp[0], hi = hp[2];
    v16bf a = __builtin_shufflevector(lo, hi, 0,1,2,3,4,5,6,7,8,9,10,11,12,13,14,15);
#pragma unroll
    for (int t = 0; t < 2; ++t) {
      v16bf b = load_frag16(W2t + (size_t)(wave*32 + t*16 + l15) * DFF_ + kb + 8*h, true);
      acc2[t] = __builtin_amdgcn_wmma_f32_16x16x32_bf16(false, a, false, b,
                                                        (short)0, acc2[t], false, false);
    }
  }
#pragma unroll
  for (int t = 0; t < 2; ++t) {
    const int col = wave*32 + t*16 + l15;
#pragma unroll
    for (int r = 0; r < 8; ++r) {
      const int row = m0 + h*8 + r;
      if (row < M)
        X[(size_t)row * D_ + col] += acc2[t][r] + b2[col];
    }
  }
}

// ---------- weight convert: W (L matrices, KxN f32) -> Wt (L matrices, NxK bf16 packed) ----------
__global__ __launch_bounds__(256) void wcvt_k(const float* __restrict__ W,
                                              unsigned* __restrict__ Wt,
                                              int K, int Nn, int total2) {
  int t = blockIdx.x*256 + threadIdx.x;
  if (t >= total2) return;
  int perM = (K >> 1) * Nn;                 // u32 per matrix
  int l = t / perM, idx = t % perM;
  int n = idx / (K >> 1), k2 = idx % (K >> 1);
  const float* Wl = W + (size_t)l * K * Nn;
  float f0 = Wl[(size_t)(2*k2)   * Nn + n];
  float f1 = Wl[(size_t)(2*k2+1) * Nn + n];
  Wt[t] = pack2bf(f0, f1);
}

// ---------- layernorm over D=256; one wave per row ----------
// variant A: writes f32 (may alias input) AND packed bf16
__global__ __launch_bounds__(256) void layernorm_f32bf(const float* __restrict__ in,
                                                       float* __restrict__ out,
                                                       unsigned* __restrict__ outb,
                                                       const float* __restrict__ g,
                                                       const float* __restrict__ b, int M) {
  const int lane = threadIdx.x & 31;
  const int wave = threadIdx.x >> 5;
  const int row  = blockIdx.x * 8 + wave;
  if (row >= M) return;
  const float* p = in + (size_t)row * D_;
  float v[8];
#pragma unroll
  for (int e = 0; e < 8; ++e) v[e] = p[lane*8 + e];
  float s = 0.f;
#pragma unroll
  for (int e = 0; e < 8; ++e) s += v[e];
  for (int off = 16; off; off >>= 1) s += __shfl_xor(s, off, 32);
  const float mu = s * (1.f / 256.f);
  float s2 = 0.f;
#pragma unroll
  for (int e = 0; e < 8; ++e) { float d = v[e] - mu; s2 += d * d; }
  for (int off = 16; off; off >>= 1) s2 += __shfl_xor(s2, off, 32);
  const float rstd = rsqrtf(s2 * (1.f / 256.f) + 1e-5f);
  float* q = out + (size_t)row * D_;
  unsigned* qb = outb + (size_t)row * (D_/2);
#pragma unroll
  for (int pp = 0; pp < 4; ++pp) {
    const int c = lane*8 + 2*pp;
    float w0 = (v[2*pp]   - mu) * rstd * g[c]   + b[c];
    float w1 = (v[2*pp+1] - mu) * rstd * g[c+1] + b[c+1];
    q[c] = w0; q[c+1] = w1;
    qb[lane*4 + pp] = pack2bf(w0, w1);
  }
}

// variant B: packed bf16 only
__global__ __launch_bounds__(256) void layernorm_bf(const float* __restrict__ in,
                                                    unsigned* __restrict__ outb,
                                                    const float* __restrict__ g,
                                                    const float* __restrict__ b, int M) {
  const int lane = threadIdx.x & 31;
  const int wave = threadIdx.x >> 5;
  const int row  = blockIdx.x * 8 + wave;
  if (row >= M) return;
  const float* p = in + (size_t)row * D_;
  float v[8];
#pragma unroll
  for (int e = 0; e < 8; ++e) v[e] = p[lane*8 + e];
  float s = 0.f;
#pragma unroll
  for (int e = 0; e < 8; ++e) s += v[e];
  for (int off = 16; off; off >>= 1) s += __shfl_xor(s, off, 32);
  const float mu = s * (1.f / 256.f);
  float s2 = 0.f;
#pragma unroll
  for (int e = 0; e < 8; ++e) { float d = v[e] - mu; s2 += d * d; }
  for (int off = 16; off; off >>= 1) s2 += __shfl_xor(s2, off, 32);
  const float rstd = rsqrtf(s2 * (1.f / 256.f) + 1e-5f);
  unsigned* qb = outb + (size_t)row * (D_/2);
#pragma unroll
  for (int pp = 0; pp < 4; ++pp) {
    const int c = lane*8 + 2*pp;
    float w0 = (v[2*pp]   - mu) * rstd * g[c]   + b[c];
    float w1 = (v[2*pp+1] - mu) * rstd * g[c+1] + b[c+1];
    qb[lane*4 + pp] = pack2bf(w0, w1);
  }
}

// ---------- fused LN + small head: out[row,c] = LN(in[row])@W + bias ----------
__global__ __launch_bounds__(256) void ln_head_k(const float* __restrict__ in,
                                                 const float* __restrict__ g,
                                                 const float* __restrict__ b,
                                                 const float* __restrict__ W, // K x Nn
                                                 const float* __restrict__ bias,
                                                 float* __restrict__ out, int M, int Nn) {
  const int lane = threadIdx.x & 31;
  const int wave = threadIdx.x >> 5;
  const int row  = blockIdx.x * 8 + wave;
  if (row >= M) return;
  const float* p = in + (size_t)row * D_;
  float v[8];
#pragma unroll
  for (int e = 0; e < 8; ++e) v[e] = p[lane*8 + e];
  float s = 0.f;
#pragma unroll
  for (int e = 0; e < 8; ++e) s += v[e];
  for (int off = 16; off; off >>= 1) s += __shfl_xor(s, off, 32);
  const float mu = s * (1.f / 256.f);
  float s2 = 0.f;
#pragma unroll
  for (int e = 0; e < 8; ++e) { float d = v[e] - mu; s2 += d * d; }
  for (int off = 16; off; off >>= 1) s2 += __shfl_xor(s2, off, 32);
  const float rstd = rsqrtf(s2 * (1.f / 256.f) + 1e-5f);
  float xn[8];
#pragma unroll
  for (int e = 0; e < 8; ++e) {
    const int c = lane*8 + e;
    xn[e] = (v[e] - mu) * rstd * g[c] + b[c];
  }
  for (int c = 0; c < Nn; ++c) {
    float acc = 0.f;
#pragma unroll
    for (int e = 0; e < 8; ++e) acc += xn[e] * W[(size_t)(lane*8 + e) * Nn + c];
    for (int off = 16; off; off >>= 1) acc += __shfl_xor(acc, off, 32);
    if (lane == 0) out[(size_t)row * Nn + c] = acc + bias[c];
  }
}

// ---------- elementwise / small kernels ----------
__global__ __launch_bounds__(256) void embed_x_k(const int* __restrict__ ids,
                                                 const float* __restrict__ emb,
                                                 float* __restrict__ x) {
  int t = blockIdx.x*256 + threadIdx.x;        // over XE_, (b,i,d)
  int d = t & 255, i = (t>>8) & 63, b = t>>14;
  x[t] = emb[(size_t)ids[b*64 + i] * D_ + d];
}

__global__ __launch_bounds__(256) void embed_e_k(const int* __restrict__ ids,
                                                 const float* __restrict__ emb,
                                                 float* __restrict__ e) {
  int t = blockIdx.x*256 + threadIdx.x;        // over EE_, (b,i,j,d)
  int d = t & 255, j = (t>>8) & 63, i = (t>>14) & 63, b = t>>20;
  e[t] = emb[(size_t)ids[(b*64 + i)*64 + j] * D_ + d];
}

// sym_tensor: for i>j, if e[b,j,i,d]!=0 copy it into e[b,i,j,d] (in-place safe)
__global__ __launch_bounds__(256) void sym_k(float* __restrict__ e) {
  int t = blockIdx.x*256 + threadIdx.x;
  int d = t & 255, j = (t>>8) & 63, i = (t>>14) & 63, b = t>>20;
  if (i > j) {
    float u = e[((((size_t)b*64 + j)*64 + i) << 8) + d];
    if (u != 0.f) e[t] = u;
  }
}

// e2 += Ni[b,i]+Nj[b,j]  (2 elems/thread), also emit packed bf16 copy
__global__ __launch_bounds__(256) void add_ninj_bf(float* __restrict__ e2,
                                                   const float* __restrict__ Ni,
                                                   const float* __restrict__ Nj,
                                                   unsigned* __restrict__ ebf) {
  int t = blockIdx.x*256 + threadIdx.x;        // over EE_/2
  int d = (t & 127) * 2, j = (t>>7) & 63, i = (t>>13) & 63, b = t>>19;
  size_t base = (size_t)t * 2;
  size_t ni = (size_t)(b*64 + i)*D_ + d;
  size_t nj = (size_t)(b*64 + j)*D_ + d;
  float a0 = e2[base]   + Ni[ni]   + Nj[nj];
  float a1 = e2[base+1] + Ni[ni+1] + Nj[nj+1];
  e2[base] = a0; e2[base+1] = a1;
  ebf[t] = pack2bf(a0, a1);
}

// att[b,i,j,h] = 0.25 * sum_d Q[b,i,h,d]*e2[b,i,j,h,d]*K[b,j,h,d]
__global__ __launch_bounds__(256) void attn_scores_k(const float* __restrict__ Q,
                                                     const float* __restrict__ Kx,
                                                     const float* __restrict__ e2,
                                                     float* __restrict__ att) {
  int t = blockIdx.x*256 + threadIdx.x;        // over ATTE_, (b,i,j,h)
  int h = t & 15, j = (t>>4) & 63, i = (t>>10) & 63, b = t>>16;
  const float* q  = Q  + ((size_t)(b*64 + i)*D_ + h*16);
  const float* k  = Kx + ((size_t)(b*64 + j)*D_ + h*16);
  const float* ee = e2 + (((size_t)(b*64 + i)*64 + j)*D_ + h*16);
  float s = 0.f;
#pragma unroll
  for (int d = 0; d < 16; ++d) s += q[d] * ee[d] * k[d];
  att[t] = s * 0.25f;                          // 1/sqrt(16)
}

// softmax over i (axis 1) for fixed (b,j,h); stride over i = N*H = 1024
__global__ __launch_bounds__(256) void softmax_i_k(float* __restrict__ att) {
  int t = blockIdx.x*256 + threadIdx.x;        // over B*N*H = 8192, (b,j,h)
  int h = t & 15, j = (t>>4) & 63, b = t>>10;
  float* base = att + (size_t)b*65536 + j*16 + h;
  float mx = -3.4e38f;
  for (int i = 0; i < 64; ++i) mx = fmaxf(mx, base[(size_t)i*1024]);
  float s = 0.f;
  for (int i = 0; i < 64; ++i) { float v = __expf(base[(size_t)i*1024] - mx);
                                 base[(size_t)i*1024] = v; s += v; }
  float inv = 1.f / s;
  for (int i = 0; i < 64; ++i) base[(size_t)i*1024] *= inv;
}

// xo[b,i,h,d] = sum_j att[b,i,j,h] * V[b,j,h,d] ; written as packed bf16 (2 d/thread)
__global__ __launch_bounds__(256) void attn_apply_bf(const float* __restrict__ att,
                                                     const float* __restrict__ V,
                                                     unsigned* __restrict__ xobf) {
  int t = blockIdx.x*256 + threadIdx.x;        // over XE_/2, (b,i,h,d2)
  int d = (t & 7) * 2, h = (t>>3) & 15, i = (t>>7) & 63, b = t>>13;
  const float* ap = att + ((size_t)(b*64 + i)*64)*16 + h;   // + j*16
  const float* vp = V + (size_t)b*64*D_ + h*16 + d;          // + j*256
  float s0 = 0.f, s1 = 0.f;
  for (int j = 0; j < 64; ++j) {
    float a = ap[(size_t)j*16];
    s0 += a * vp[(size_t)j*D_];
    s1 += a * vp[(size_t)j*D_ + 1];
  }
  xobf[t] = pack2bf(s0, s1);
}

__global__ __launch_bounds__(256) void mean_nodes_bf(const float* __restrict__ x,
                                                     unsigned* __restrict__ molbf) {
  int t = blockIdx.x*256 + threadIdx.x;        // over B*D/2 = 1024
  int d = (t & 127) * 2, b = t >> 7;
  const float* p = x + (size_t)b*64*D_ + d;
  float s0 = 0.f, s1 = 0.f;
  for (int i = 0; i < 64; ++i) { s0 += p[(size_t)i*D_]; s1 += p[(size_t)i*D_ + 1]; }
  molbf[t] = pack2bf(s0 * (1.f/64.f), s1 * (1.f/64.f));
}

__global__ __launch_bounds__(256) void z_bf(const float* __restrict__ qmu,
                                            const float* __restrict__ qlv,
                                            const float* __restrict__ eps,
                                            unsigned* __restrict__ zbf) {
  int t = blockIdx.x*256 + threadIdx.x;        // over B*DZ/2 = 256
  if (t >= (B_*DZ_)/2) return;
  float z0 = qmu[2*t]   + eps[2*t]   * __expf(0.5f * qlv[2*t]);
  float z1 = qmu[2*t+1] + eps[2*t+1] * __expf(0.5f * qlv[2*t+1]);
  zbf[t] = pack2bf(z0, z1);
}

__global__ __launch_bounds__(256) void dec_init_x_k(const float* __restrict__ zp,
                                                    const float* __restrict__ pe,
                                                    float* __restrict__ x) {
  int t = blockIdx.x*256 + threadIdx.x;        // over XE_
  int d = t & 255, i = (t>>8) & 63, b = t>>14;
  x[t] = zp[(size_t)b*D_ + d] + pe[(size_t)i*D_ + d];
}

// e[b,i,j,d] = z[b,d] + pe[i,d] + pe[j,d]  (both sym_tensor calls are identities)
__global__ __launch_bounds__(256) void dec_init_e_k(const float* __restrict__ zp,
                                                    const float* __restrict__ pe,
                                                    float* __restrict__ e) {
  int t = blockIdx.x*256 + threadIdx.x;        // over EE_
  int d = t & 255, j = (t>>8) & 63, i = (t>>14) & 63, b = t>>20;
  e[t] = zp[(size_t)b*D_ + d] + pe[(size_t)i*D_ + d] + pe[(size_t)j*D_ + d];
}

// ---------- host orchestration ----------
// Logical parameter order (as created in stack()):
enum { pWq, pbq, pWk, pbk, pWv, pbv, pWe, pbe, pWni, pbni, pWnj, pbnj,
       pWOx, pbOx, pWOe, pbOe, pW1x, pb1x, pW2x, pb2x, pW1e, pb1e, pW2e, pb2e,
       pg_lnx, pb_lnx, pg_lne, pb_lne, pg_lnx2, pb_lnx2, pg_lne2, pb_lne2 };

// JAX sorted-key flatten order: logical idx -> position
static const int kSortedPos[32] = {
  10, 26, 7, 23, 11, 27, 6, 22, 8, 24, 9, 25, 5, 17, 4, 16,
  1, 13, 3, 15, 0, 12, 2, 14, 30, 20, 28, 18, 31, 21, 29, 19 };

static const size_t kPerL[32] = {
  65536,256, 65536,256, 65536,256, 65536,256, 65536,256, 65536,256, 65536,256, 65536,256,
  262144,1024, 262144,256, 262144,1024, 262144,256,
  256,256, 256,256, 256,256, 256,256 };

// bf16 weight arena: per-stack leaf list (logical ids) with K,N
static const int kWLeaf[12] = { pWq, pWk, pWv, pWe, pWni, pWnj, pWOx, pWOe,
                                pW1x, pW2x, pW1e, pW2e };
static const int kWK[12] = { 256,256,256,256,256,256,256,256, 256,1024,256,1024 };
static const int kWN[12] = { 256,256,256,256,256,256,256,256, 1024,256,1024,256 };

extern "C" void kernel_launch(void* const* d_in, const int* in_sizes, int n_in,
                              void* d_out, int out_size, void* d_ws, size_t ws_size,
                              hipStream_t stream) {
  if (n_in < 84) return;

  const int*   x_ids    = (const int*)d_in[0];
  const int*   e_ids    = (const int*)d_in[1];
  const float* eps      = (const float*)d_in[2];
  const float* atom_emb = (const float*)d_in[3];
  const float* bond_emb = (const float*)d_in[4];
  const float* pe       = (const float*)d_in[5];
  const float* w_qmu = (const float*)d_in[70]; const float* b_qmu = (const float*)d_in[71];
  const float* w_qlv = (const float*)d_in[72]; const float* b_qlv = (const float*)d_in[73];
  const float* w_p   = (const float*)d_in[74]; const float* b_p   = (const float*)d_in[75];
  const float* lnxf_g= (const float*)d_in[76]; const float* lnxf_b= (const float*)d_in[77];
  const float* w_xf  = (const float*)d_in[78]; const float* b_xf  = (const float*)d_in[79];
  const float* lnef_g= (const float*)d_in[80]; const float* lnef_b= (const float*)d_in[81];
  const float* w_ef  = (const float*)d_in[82]; const float* b_ef  = (const float*)d_in[83];

  // detect dict-flatten order: sorted keys put W1e (L*D*DFF = 1,048,576) first
  const bool sortedKeys = (in_sizes[6] == (int)(L_ * (size_t)D_ * DFF_));

  // ---- workspace carve-up (all 4-byte words; every bf16 buffer 16B aligned) ----
  float* ws = (float*)d_ws;
  size_t off = 0;
  auto allocF = [&](size_t n) { float* p = ws + off; off += n; return p; };
  float* e   = allocF(EE_);
  float* e2  = allocF(EE_);
  float* att = allocF(ATTE_);
  float* x   = allocF(XE_);
  float* xq  = allocF(XE_);
  float* xk  = allocF(XE_);
  float* xv  = allocF(XE_);
  float* xni = allocF(XE_);
  float* xnj = allocF(XE_);
  float* zp  = allocF(B_*D_);
  auto allocB = [&](size_t nbf16) { unsigned* p = (unsigned*)(ws + off); off += nbf16/2; return p; };
  unsigned* ebf   = allocB(EE_);
  unsigned* xbf   = allocB(XE_);
  unsigned* xobf  = allocB(XE_);
  unsigned* molbf = allocB(B_*D_);
  unsigned* zbf   = allocB(B_*DZ_);
  // weight arena: 2 stacks x 12 leaves + 3 latent
  unsigned* wb[2][12];
  for (int s = 0; s < 2; ++s)
    for (int w = 0; w < 12; ++w)
      wb[s][w] = allocB((size_t)L_ * kWK[w] * kWN[w]);
  unsigned* wqmut = allocB((size_t)D_ * DZ_);
  unsigned* wqlvt = allocB((size_t)D_ * DZ_);
  unsigned* wpt   = allocB((size_t)DZ_ * D_);
  if (ws_size < off * sizeof(float)) return;

  float* out   = (float*)d_out;
  float* o_x   = out;                       // (B,N,NA)  = 8192
  float* o_e   = out + 8192;                // (B,N,N,NB)= 262144
  float* o_qmu = out + 8192 + 262144;       // (B,DZ)    = 512
  float* o_qlv = o_qmu + 512;               // (B,DZ)    = 512
  (void)out_size;

  auto leafPtr = [&](int base, int logical) -> const float* {
    int pos = sortedKeys ? kSortedPos[logical] : logical;
    return (const float*)d_in[base + pos];
  };

  // ---- one-time weight conversion (transposed bf16) ----
  auto wcvt = [&](const float* W, unsigned* dst, int K, int Nn, int Lcount) {
    int total2 = Lcount * (K/2) * Nn;
    wcvt_k<<<(total2 + 255)/256, 256, 0, stream>>>(W, dst, K, Nn, total2);
  };
  for (int s = 0; s < 2; ++s) {
    const int base = (s == 0) ? 6 : 38;
    for (int w = 0; w < 12; ++w)
      wcvt(leafPtr(base, kWLeaf[w]), wb[s][w], kWK[w], kWN[w], L_);
  }
  wcvt(w_qmu, wqmut, D_, DZ_, 1);
  wcvt(w_qlv, wqlvt, D_, DZ_, 1);
  wcvt(w_p,   wpt,   DZ_, D_, 1);

  auto GEMM = [&](const unsigned* Ab, const unsigned* Wt, const float* bias, const float* R,
                  float* C, int M, int K, int Nn, bool resid) {
    dim3 grid((Nn + 63) / 64, (M + 63) / 64);
    if (resid) gemm_bf16t<true ><<<grid, 128, 0, stream>>>((const __bf16*)Ab, (const __bf16*)Wt,
                                                           bias, R, C, M, K, Nn);
    else       gemm_bf16t<false><<<grid, 128, 0, stream>>>((const __bf16*)Ab, (const __bf16*)Wt,
                                                           bias, nullptr, C, M, K, Nn);
  };

  auto runStack = [&](int s) {
    const int base = (s == 0) ? 6 : 38;
    for (int l = 0; l < L_; ++l) {
      auto P = [&](int logical) -> const float* {
        return leafPtr(base, logical) + (size_t)l * kPerL[logical];
      };
      auto WT = [&](int w) -> const unsigned* {          // per-layer transposed bf16 weight
        return wb[s][w] + (size_t)l * kWK[w] * kWN[w] / 2;
      };
      // pre-norm (reference overwrites x/e with their normalized versions)
      layernorm_f32bf<<<BN_/8, 256, 0, stream>>>(x, x, xbf, P(pg_lnx), P(pb_lnx), BN_);
      layernorm_f32bf<<<BNN_/8, 256, 0, stream>>>(e, e, ebf, P(pg_lne), P(pb_lne), BNN_);
      // node projections (all read xbf)
      GEMM(xbf, WT(0), P(pbq),  nullptr, xq,  BN_, D_, D_, false);
      GEMM(xbf, WT(1), P(pbk),  nullptr, xk,  BN_, D_, D_, false);
      GEMM(xbf, WT(2), P(pbv),  nullptr, xv,  BN_, D_, D_, false);
      GEMM(xbf, WT(4), P(pbni), nullptr, xni, BN_, D_, D_, false);
      GEMM(xbf, WT(5), P(pbnj), nullptr, xnj, BN_, D_, D_, false);
      // edge projection -> e2, then += Ni/Nj broadcast (also emits ebf = bf16(e2))
      GEMM(ebf, WT(3), P(pbe), nullptr, e2, BNN_, D_, D_, false);
      add_ninj_bf<<<(int)(EE_/2/256), 256, 0, stream>>>(e2, xni, xnj, ebf);
      // attention
      attn_scores_k<<<ATTE_/256, 256, 0, stream>>>(xq, xk, e2, att);
      softmax_i_k<<<(B_*N_*H_)/256, 256, 0, stream>>>(att);
      attn_apply_bf<<<XE_/2/256, 256, 0, stream>>>(att, xv, xobf);
      // output projections with in-place residual
      GEMM(xobf, WT(6), P(pbOx), x, x, BN_,  D_, D_, true);
      GEMM(ebf,  WT(7), P(pbOe), e, e, BNN_, D_, D_, true);
      // FFNs (fused; residual accumulated in place)
      layernorm_bf<<<BN_/8, 256, 0, stream>>>(x, xbf, P(pg_lnx2), P(pb_lnx2), BN_);
      ffn_fused_bf<<<BN_/16, 256, 0, stream>>>((const __bf16*)xbf, (const __bf16*)wb[s][8] + (size_t)l*D_*DFF_,
                                               P(pb1x), (const __bf16*)wb[s][9] + (size_t)l*DFF_*D_,
                                               P(pb2x), x, BN_);
      layernorm_bf<<<BNN_/8, 256, 0, stream>>>(e, ebf, P(pg_lne2), P(pb_lne2), BNN_);
      ffn_fused_bf<<<BNN_/16, 256, 0, stream>>>((const __bf16*)ebf, (const __bf16*)wb[s][10] + (size_t)l*D_*DFF_,
                                                P(pb1e), (const __bf16*)wb[s][11] + (size_t)l*DFF_*D_,
                                                P(pb2e), e, BNN_);
      // symmetrize after every block
      sym_k<<<(int)(EE_/256), 256, 0, stream>>>(e);
    }
  };

  // ---- encoder ----
  embed_x_k<<<XE_/256, 256, 0, stream>>>(x_ids, atom_emb, x);
  embed_e_k<<<(int)(EE_/256), 256, 0, stream>>>(e_ids, bond_emb, e);
  sym_k<<<(int)(EE_/256), 256, 0, stream>>>(e);
  runStack(0);

  // ---- latent ----
  mean_nodes_bf<<<(B_*D_/2)/256, 256, 0, stream>>>(x, molbf);
  GEMM(molbf, wqmut, b_qmu, nullptr, o_qmu, B_, D_, DZ_, false);
  GEMM(molbf, wqlvt, b_qlv, nullptr, o_qlv, B_, D_, DZ_, false);
  z_bf<<<1, 256, 0, stream>>>(o_qmu, o_qlv, eps, zbf);
  GEMM(zbf, wpt, b_p, nullptr, zp, B_, DZ_, D_, false);

  // ---- decoder init (sym_tensor calls are identities on symmetric inputs) ----
  dec_init_x_k<<<XE_/256, 256, 0, stream>>>(zp, pe, x);
  dec_init_e_k<<<(int)(EE_/256), 256, 0, stream>>>(zp, pe, e);
  runStack(1);

  // ---- output heads (fused LN + projection) ----
  ln_head_k<<<BN_/8, 256, 0, stream>>>(x, lnxf_g, lnxf_b, w_xf, b_xf, o_x, BN_, 16);
  ln_head_k<<<BNN_/8, 256, 0, stream>>>(e, lnef_g, lnef_b, w_ef, b_ef, o_e, BNN_, 8);
}
